// SequenceMemoryUpdater_80049600463361
// MI455X (gfx1250) — matrix operand order repaired and checked by reading, not compiled, not run on hardware
//
#include <hip/hip_runtime.h>
#include <hip/hip_bf16.h>

// Fixed problem geometry from the reference: H=128, M=256, 3H=384.
#define HDIM   128
#define MDIM   256
#define G3     384
#define ROWS   64              // rows (updated nodes) per block
#define NT     24              // 384/16 column tiles
#define KT_IH  8               // 256/32 k-steps for W_ih
#define KT_HH  4               // 128/32 k-steps for W_hh
#define SA_MSG 264             // padded bf16 row stride for staged messages
#define SA_H   136             // padded bf16 row stride for staged h
#define GS     388             // padded f32 row stride for gate buffers

#define IH_ELEMS (NT * KT_IH * 32 * 16)   // 98304 bf16
#define HH_ELEMS (NT * KT_HH * 32 * 16)   // 49152 bf16
#define WS_ELEMS (IH_ELEMS + HH_ELEMS)    // 147456 bf16 = 294912 bytes

typedef __bf16 bf16x8  __attribute__((ext_vector_type(8)));
typedef __bf16 v16bf   __attribute__((ext_vector_type(16)));
typedef float  v8f     __attribute__((ext_vector_type(8)));

__device__ __forceinline__ unsigned short f2bf(float f) {
    unsigned int u = __float_as_uint(f);
    // round-to-nearest-even bf16
    unsigned int r = (u + 0x7fffu + ((u >> 16) & 1u)) >> 16;
    return (unsigned short)r;
}

// ---------------------------------------------------------------------------
// Prepack W_ih [384,256] and W_hh [384,128] (f32, row-major) into bf16
// B-fragment order for v_wmma_f32_16x16x32_bf16:
//   fragment(nt, kt): lane l -> column n = l&15, half = l>>4,
//   element i (0..15) -> K = kt*32 + half*16 + i ;  B[k][n] = W[nt*16+n][k]
// Flat index: ((nt*KT + kt)*32 + lane)*16 + i
// ---------------------------------------------------------------------------
__global__ __launch_bounds__(256) void prepack_weights(
        const float* __restrict__ Wih, const float* __restrict__ Whh,
        unsigned short* __restrict__ ws) {
    int idx = blockIdx.x * 256 + threadIdx.x;
    if (idx >= WS_ELEMS) return;
    float v;
    if (idx < IH_ELEMS) {
        int i16  = idx & 15;
        int lane = (idx >> 4) & 31;
        int kt   = (idx >> 9) & 7;
        int nt   = idx >> 12;
        int j = nt * 16 + (lane & 15);
        int k = kt * 32 + (lane >> 4) * 16 + i16;
        v = Wih[j * MDIM + k];
    } else {
        int e    = idx - IH_ELEMS;
        int i16  = e & 15;
        int lane = (e >> 4) & 31;
        int kt   = (e >> 9) & 3;
        int nt   = e >> 11;
        int j = nt * 16 + (lane & 15);
        int k = kt * 32 + (lane >> 4) * 16 + i16;
        v = Whh[j * HDIM + k];
    }
    ws[idx] = f2bf(v);
}

// ---------------------------------------------------------------------------
// One 16-col n-tile x 64 rows GEMM job: 4 M-tile accumulators share each B
// fragment. KT / SA are compile-time so all loop control stays scalar and
// EXEC remains all-ones around every v_wmma (ISA requirement).
// ---------------------------------------------------------------------------
template <int KT, int SA>
__device__ __forceinline__ void gemm_tile(
        const unsigned short* __restrict__ asrc,   // LDS bf16, padded rows
        const unsigned short* __restrict__ bbase,  // global prepacked B frags
        float* __restrict__ g,                     // LDS f32 gate buffer
        int nt, int lane, int col, int half) {
    v8f acc[4];
    #pragma unroll
    for (int mt = 0; mt < 4; ++mt) acc[mt] = v8f{};

    for (int kt = 0; kt < KT; ++kt) {
        // B fragment: 16 contiguous bf16 per lane (32B, coalesced, L2-hot)
        v16bf b = *(const v16bf*)(bbase + ((size_t)kt * 32 + lane) * 16);
        #pragma unroll
        for (int mt = 0; mt < 4; ++mt) {
            // A fragment: two 16B groups per lane from padded LDS rows
            const unsigned short* ap =
                asrc + (mt * 16 + col) * SA + kt * 32 + half * 16;
            bf16x8 g1 = *(const bf16x8*)(ap);
            bf16x8 g2 = *(const bf16x8*)(ap + 8);
            v16bf a = __builtin_shufflevector(
                g1, g2, 0, 1, 2, 3, 4, 5, 6, 7,
                8, 9, 10, 11, 12, 13, 14, 15);
            acc[mt] = __builtin_amdgcn_wmma_f32_16x16x32_bf16(
                false, a, false, b, (short)0, acc[mt], false, false);
        }
    }

    const int n = nt * 16 + col;
    #pragma unroll
    for (int mt = 0; mt < 4; ++mt) {
        const int mrow = mt * 16 + half * 8;
        #pragma unroll
        for (int v = 0; v < 8; ++v)
            g[(mrow + v) * GS + n] = acc[mt][v];
    }
}

// ---------------------------------------------------------------------------
// Fused GRU memory-update kernel: gather h, two WMMA GEMMs into LDS gate
// buffers, element-wise GRU, masked scatter into the (pre-copied) output.
// ---------------------------------------------------------------------------
__global__ __launch_bounds__(256) void gru_update(
        const int*   __restrict__ node_ids,
        const float* __restrict__ msgs,
        const float* __restrict__ ts,
        const float* __restrict__ mem,
        const float* __restrict__ lastu,
        const float* __restrict__ b_ih,
        const float* __restrict__ b_hh,
        const unsigned short* __restrict__ wpack,
        float* __restrict__ outMem,
        float* __restrict__ outLU,
        int nU) {
    __shared__ float          sGi[ROWS * GS];        //  99328 B
    __shared__ float          sGh[ROWS * GS];        //  99328 B
    __shared__ unsigned short sMsg[ROWS * SA_MSG];   //  33792 B
    __shared__ unsigned short sH[ROWS * SA_H];       //  17408 B
    __shared__ float          sBias[2 * G3];         //   3072 B
    __shared__ int            sNode[ROWS];
    __shared__ int            sMask[ROWS];

    const int tid = threadIdx.x;
    const int r0  = blockIdx.x * ROWS;

    // ---- stage node ids / timestamp mask / biases --------------------------
    if (tid < ROWS) {
        int r = r0 + tid;
        int nd = 0, mk = 0;
        if (r < nU) {
            nd = node_ids[r];
            mk = (lastu[nd] <= ts[r]) ? 1 : 0;
        }
        sNode[tid] = nd;
        sMask[tid] = mk;
    }
    for (int i = tid; i < 2 * G3; i += 256)
        sBias[i] = (i < G3) ? b_ih[i] : b_hh[i - G3];
    __syncthreads();

    // ---- stage A operands (messages, gathered h) as bf16 in LDS ------------
    for (int i = tid; i < ROWS * MDIM; i += 256) {
        int m = i >> 8, k = i & 255;
        int r = r0 + m;
        float v = (r < nU) ? msgs[(size_t)r * MDIM + k] : 0.f;
        sMsg[m * SA_MSG + k] = f2bf(v);
    }
    for (int i = tid; i < ROWS * HDIM; i += 256) {
        int m = i >> 7, k = i & 127;
        int r = r0 + m;
        float v = (r < nU) ? mem[(size_t)sNode[m] * HDIM + k] : 0.f;
        sH[m * SA_H + k] = f2bf(v);
    }
    __syncthreads();

    // ---- GEMM phase --------------------------------------------------------
    // Scalarized wave id -> all GEMM loop control is SALU, EXEC stays all-1s.
    const int wv   = __builtin_amdgcn_readfirstlane(tid >> 5);
    const int lane = tid & 31;
    const int half = lane >> 4;
    const int col  = lane & 15;

    // gi = messages @ W_ih^T   (K = 256)
    for (int nt = wv; nt < NT; nt += 8)
        gemm_tile<KT_IH, SA_MSG>(
            sMsg, wpack + (size_t)nt * (KT_IH * 32 * 16),
            sGi, nt, lane, col, half);

    // gh = h @ W_hh^T          (K = 128)
    for (int nt = wv; nt < NT; nt += 8)
        gemm_tile<KT_HH, SA_H>(
            sH, wpack + (size_t)IH_ELEMS + (size_t)nt * (KT_HH * 32 * 16),
            sGh, nt, lane, col, half);

    __syncthreads();

    // ---- element-wise GRU + masked scatter ---------------------------------
    for (int i = tid; i < ROWS * HDIM; i += 256) {
        int m = i >> 7, j = i & 127;
        if (!sMask[m]) continue;
        const int base = m * GS;
        float rr = sGi[base + j] + sGh[base + j] + sBias[j] + sBias[G3 + j];
        rr = 1.f / (1.f + __expf(-rr));
        float zz = sGi[base + 128 + j] + sGh[base + 128 + j] +
                   sBias[128 + j] + sBias[G3 + 128 + j];
        zz = 1.f / (1.f + __expf(-zz));
        float nn = sGi[base + 256 + j] + sBias[256 + j] +
                   rr * (sGh[base + 256 + j] + sBias[G3 + 256 + j]);
        nn = tanhf(nn);
        size_t o = (size_t)sNode[m] * HDIM + j;
        float hv = mem[o];                    // exact f32 h (L2-hot)
        outMem[o] = (1.f - zz) * nn + zz * hv;
    }
    if (tid < ROWS) {
        int r = r0 + tid;
        if (r < nU && sMask[tid]) outLU[sNode[tid]] = ts[r];
    }
}

// ---------------------------------------------------------------------------
extern "C" void kernel_launch(void* const* d_in, const int* in_sizes, int n_in,
                              void* d_out, int out_size, void* d_ws, size_t ws_size,
                              hipStream_t stream) {
    const int*   node_ids  = (const int*)d_in[0];
    const float* messages  = (const float*)d_in[1];
    const float* timestamps= (const float*)d_in[2];
    const float* memory    = (const float*)d_in[3];
    const float* last_upd  = (const float*)d_in[4];
    const float* W_ih      = (const float*)d_in[5];
    const float* W_hh      = (const float*)d_in[6];
    const float* b_ih      = (const float*)d_in[7];
    const float* b_hh      = (const float*)d_in[8];

    const int nU     = in_sizes[0];
    const int nNodes = in_sizes[3] / HDIM;

    float* outMem = (float*)d_out;
    float* outLU  = outMem + (size_t)nNodes * HDIM;

    // Base output = unmodified memory / last_update (masked rows stay as-is).
    hipMemcpyAsync(outMem, memory, (size_t)nNodes * HDIM * sizeof(float),
                   hipMemcpyDeviceToDevice, stream);
    hipMemcpyAsync(outLU, last_upd, (size_t)nNodes * sizeof(float),
                   hipMemcpyDeviceToDevice, stream);

    if (nU <= 0) return;
    unsigned short* wpack = (unsigned short*)d_ws;   // needs 294912 B

    prepack_weights<<<(WS_ELEMS + 255) / 256, 256, 0, stream>>>(W_ih, W_hh, wpack);

    const int nBlocks = (nU + ROWS - 1) / ROWS;
    gru_update<<<nBlocks, 256, 0, stream>>>(
        node_ids, messages, timestamps, memory, last_upd,
        b_ih, b_hh, wpack, outMem, outLU, nU);
}
